// MultiHead_91207925498550
// MI455X (gfx1250) — matrix operand order repaired and checked by reading, not compiled, not run on hardware
//
#include <hip/hip_runtime.h>

typedef _Float16 f16;
typedef __attribute__((ext_vector_type(16))) f16   v16h;
typedef __attribute__((ext_vector_type(8)))  f16   v8h;
typedef __attribute__((ext_vector_type(8)))  float v8f;

#define DEV __device__ __forceinline__

// ---------------------------------------------------------------- helpers

DEV f16 f2h(float f) { return (f16)f; }

// A-matrix fragment 16x32 fp16 from row-major [.. , stride] memory.
// ISA 7.12.2: lane l -> row l%16; elems 0..7 = K 8*(l>>4)+0..7 ; elems 8..15 = K 16+8*(l>>4)+0..7
DEV v16h load_a(const f16* __restrict__ base, int stride, int row0, int k0, int lane) {
  const int m = lane & 15, half = lane >> 4;
  const f16* p = base + (size_t)(row0 + m) * stride + k0 + 8 * half;
  v8h lo = *(const v8h*)p;          // K = 8*half + 0..7
  v8h hi = *(const v8h*)(p + 16);   // K = 16 + 8*half + 0..7
  v16h r;
#pragma unroll
  for (int i = 0; i < 8; ++i) { r[i] = lo[i]; r[8 + i] = hi[i]; }
  return r;
}

// B-matrix fragment 32x16 fp16, from *transposed* storage BT[n][k] (row-major, stride).
// lane l -> column n = l%16 ; elems 0..7 = K 8*(l>>4)+0..7 ; elems 8..15 = K 16+8*(l>>4)+0..7
DEV v16h load_b(const f16* __restrict__ base, int stride, int n0, int k0, int lane) {
  const int n = lane & 15, half = lane >> 4;
  const f16* p = base + (size_t)(n0 + n) * stride + k0 + 8 * half;
  v8h lo = *(const v8h*)p;
  v8h hi = *(const v8h*)(p + 16);
  v16h r;
#pragma unroll
  for (int i = 0; i < 8; ++i) { r[i] = lo[i]; r[8 + i] = hi[i]; }
  return r;
}

DEV v8f wmma16(v16h a, v16h b, v8f c) {
  return __builtin_amdgcn_wmma_f32_16x16x32_f16(false, a, false, b, (short)0, c, false, false);
}

// ---------------------------------------------------------------- prep kernels

__global__ void k_cvt_f16(const float* __restrict__ s, f16* __restrict__ d, int n) {
  for (int i = blockIdx.x * blockDim.x + threadIdx.x; i < n; i += gridDim.x * blockDim.x)
    d[i] = f2h(s[i]);
}

// src [rows][cols] fp32  ->  dst [cols][rows] fp16
__global__ void k_transpose_f16(const float* __restrict__ s, f16* __restrict__ d,
                                int rows, int cols) {
  const int n = rows * cols;
  for (int i = blockIdx.x * blockDim.x + threadIdx.x; i < n; i += gridDim.x * blockDim.x) {
    int r = i / cols, c = i % cols;
    d[(size_t)c * rows + r] = f2h(s[i]);
  }
}

__global__ void k_zero32(unsigned int* __restrict__ d, int n) {
  for (int i = blockIdx.x * blockDim.x + threadIdx.x; i < n; i += gridDim.x * blockDim.x)
    d[i] = 0u;
}

// ---------------------------------------------------------------- QKV GEMM
// x16 [8192][768] @ W_attn -> qkv [8192][2304], using WaT [2304][768] (= W^T fp16).
// Scatter epilogue: q -> [B*H][T][64] (scaled by 1/sqrt(48)), k -> [B*H][T][64],
// v -> transposed [B*H][64][T]. Pads d=48..63 of q/k pre-zeroed.
__global__ __launch_bounds__(256) void qkv_gemm(
    const f16* __restrict__ xb, const f16* __restrict__ WaT,
    f16* __restrict__ qb, f16* __restrict__ kb, f16* __restrict__ vT) {
  const int lane = threadIdx.x & 31;
  const int wid  = threadIdx.x >> 5;
  const int wm = wid & 3, wn = wid >> 2;               // 4 x 2 wave grid
  const int row0 = blockIdx.y * 128 + wm * 32;
  const int col0 = blockIdx.x * 128 + wn * 64;

  v8f z = {};
  v8f acc[2][4];
#pragma unroll
  for (int i = 0; i < 2; ++i)
#pragma unroll
    for (int j = 0; j < 4; ++j) acc[i][j] = z;

  for (int k0 = 0; k0 < 768; k0 += 32) {
    v16h a0 = load_a(xb, 768, row0,      k0, lane);
    v16h a1 = load_a(xb, 768, row0 + 16, k0, lane);
#pragma unroll
    for (int j = 0; j < 4; ++j) {
      v16h b = load_b(WaT, 768, col0 + 16 * j, k0, lane);
      acc[0][j] = wmma16(a0, b, acc[0][j]);
      acc[1][j] = wmma16(a1, b, acc[1][j]);
    }
  }

  const float QSCALE = 0.14433756729740643f;  // 48^-0.5 folded into Q
  const int half = lane >> 4, nn = lane & 15;
#pragma unroll
  for (int i = 0; i < 2; ++i)
#pragma unroll
    for (int j = 0; j < 4; ++j) {
      const int col = col0 + 16 * j + nn;
#pragma unroll
      for (int e = 0; e < 8; ++e) {
        const int row = row0 + 16 * i + 8 * half + e;
        const int bi = row >> 11, t = row & 2047;
        const float v = acc[i][j][e];
        if (col < 768) {
          const int h = col / 48, d = col % 48;
          qb[(((size_t)(bi * 16 + h)) * 2048 + t) * 64 + d] = f2h(v * QSCALE);
        } else if (col < 1536) {
          const int c = col - 768, h = c / 48, d = c % 48;
          kb[(((size_t)(bi * 16 + h)) * 2048 + t) * 64 + d] = f2h(v);
        } else {
          const int c = col - 1536, h = c / 48, d = c % 48;
          vT[(((size_t)(bi * 16 + h)) * 64 + d) * 2048 + t] = f2h(v);
        }
      }
    }
}

// ---------------------------------------------------------------- flash attention
// One wave owns 16 queries; block = 8 waves = 128 queries; grid (T/128, B*H).
// Streams keys in blocks of 32 with online softmax; O accumulated fp32 (3 d-tiles).
__global__ __launch_bounds__(256) void attn_kernel(
    const f16* __restrict__ qb, const f16* __restrict__ kbuf,
    const f16* __restrict__ vT, f16* __restrict__ ab) {
  __shared__ f16 pbuf[8][16 * 32];   // per-wave P staging (C-layout -> A-layout)
  const int lane = threadIdx.x & 31;
  const int wid  = threadIdx.x >> 5;
  const int bh = blockIdx.y;
  const int b = bh >> 4, h = bh & 15;
  const f16* qh = qb   + (size_t)bh * (2048 * 64);
  const f16* kh = kbuf + (size_t)bh * (2048 * 64);
  const f16* vh = vT   + (size_t)bh * (64 * 2048);
  const int q0 = blockIdx.x * 128 + wid * 16;
  const int half = lane >> 4, nn = lane & 15;

  const v16h qa0 = load_a(qh, 64, q0, 0, lane);
  const v16h qa1 = load_a(qh, 64, q0, 32, lane);
  v8f z = {};
  v8f o0 = z, o1 = z, o2 = z;
  float rmax[8], rsum[8];
#pragma unroll
  for (int j = 0; j < 8; ++j) { rmax[j] = -3.0e38f; rsum[j] = 0.0f; }
  f16* pw = &pbuf[wid][0];

  const int nkb = (q0 + 47) >> 5;                 // causal: keys <= q0+15
  for (int kb2 = 0; kb2 < nkb; ++kb2) {
    const int kbase = kb2 * 32;
    v8f s0 = z, s1 = z;
    {
      v16h bk = load_b(kh, 64, kbase, 0, lane);       s0 = wmma16(qa0, bk, s0);
      bk      = load_b(kh, 64, kbase, 32, lane);      s0 = wmma16(qa1, bk, s0);
      bk      = load_b(kh, 64, kbase + 16, 0, lane);  s1 = wmma16(qa0, bk, s1);
      bk      = load_b(kh, 64, kbase + 16, 32, lane); s1 = wmma16(qa1, bk, s1);
    }
    const int key0 = kbase + nn, key1 = kbase + 16 + nn;
#pragma unroll
    for (int j = 0; j < 8; ++j) {
      const int qrow = q0 + 8 * half + j;
      float v0 = (key0 <= qrow) ? s0[j] : -3.0e38f;
      float v1 = (key1 <= qrow) ? s1[j] : -3.0e38f;
      float c = fmaxf(v0, v1);                         // row max over 16 lanes
      c = fmaxf(c, __shfl_xor(c, 1, 16));
      c = fmaxf(c, __shfl_xor(c, 2, 16));
      c = fmaxf(c, __shfl_xor(c, 4, 16));
      c = fmaxf(c, __shfl_xor(c, 8, 16));
      const float mnew = fmaxf(rmax[j], c);
      const float sc = __expf(rmax[j] - mnew);
      const float e0 = __expf(v0 - mnew);
      const float e1 = __expf(v1 - mnew);
      float ps = e0 + e1;
      ps += __shfl_xor(ps, 1, 16);
      ps += __shfl_xor(ps, 2, 16);
      ps += __shfl_xor(ps, 4, 16);
      ps += __shfl_xor(ps, 8, 16);
      rsum[j] = rsum[j] * sc + ps;
      rmax[j] = mnew;
      o0[j] *= sc; o1[j] *= sc; o2[j] *= sc;
      pw[(8 * half + j) * 32 + nn]      = f2h(e0);
      pw[(8 * half + j) * 32 + 16 + nn] = f2h(e1);
    }
    __builtin_amdgcn_fence(__ATOMIC_ACQ_REL, "wavefront");   // order DS write->read in-wave
    const v16h pa = load_a(pw, 32, 0, 0, lane);              // P as A-fragment (16x32)
    {
      v16h bv = load_b(vh, 2048, 0,  kbase, lane); o0 = wmma16(pa, bv, o0);
      bv      = load_b(vh, 2048, 16, kbase, lane); o1 = wmma16(pa, bv, o1);
      bv      = load_b(vh, 2048, 32, kbase, lane); o2 = wmma16(pa, bv, o2);
    }
  }

#pragma unroll
  for (int j = 0; j < 8; ++j) {
    const int t = q0 + 8 * half + j;
    const float inv = 1.0f / rsum[j];
    const size_t off = ((size_t)(b * 2048 + t)) * 768 + h * 48 + nn;
    ab[off]      = f2h(o0[j] * inv);
    ab[off + 16] = f2h(o1[j] * inv);
    ab[off + 32] = f2h(o2[j] * inv);
  }
}

// ---------------------------------------------------------------- proj GEMM
// out fp32 [8192][768] = ab [8192][768] @ W_proj + b_proj, via WpT [768][768].
__global__ __launch_bounds__(256) void proj_gemm(
    const f16* __restrict__ ab, const f16* __restrict__ WpT,
    const float* __restrict__ bias, float* __restrict__ out) {
  const int lane = threadIdx.x & 31;
  const int wid  = threadIdx.x >> 5;
  const int wm = wid & 3, wn = wid >> 2;
  const int row0 = blockIdx.y * 128 + wm * 32;
  const int col0 = blockIdx.x * 128 + wn * 64;

  v8f z = {};
  v8f acc[2][4];
#pragma unroll
  for (int i = 0; i < 2; ++i)
#pragma unroll
    for (int j = 0; j < 4; ++j) acc[i][j] = z;

  for (int k0 = 0; k0 < 768; k0 += 32) {
    v16h a0 = load_a(ab, 768, row0,      k0, lane);
    v16h a1 = load_a(ab, 768, row0 + 16, k0, lane);
#pragma unroll
    for (int j = 0; j < 4; ++j) {
      v16h b = load_b(WpT, 768, col0 + 16 * j, k0, lane);
      acc[0][j] = wmma16(a0, b, acc[0][j]);
      acc[1][j] = wmma16(a1, b, acc[1][j]);
    }
  }

  const int half = lane >> 4, nn = lane & 15;
#pragma unroll
  for (int i = 0; i < 2; ++i)
#pragma unroll
    for (int j = 0; j < 4; ++j) {
      const int col = col0 + 16 * j + nn;
      const float bv = bias[col];
#pragma unroll
      for (int e = 0; e < 8; ++e) {
        const int row = row0 + 16 * i + 8 * half + e;
        out[(size_t)row * 768 + col] = acc[i][j][e] + bv;
      }
    }
}

// ---------------------------------------------------------------- launch

extern "C" void kernel_launch(void* const* d_in, const int* in_sizes, int n_in,
                              void* d_out, int out_size, void* d_ws, size_t ws_size,
                              hipStream_t stream) {
  const float* x      = (const float*)d_in[0];   // [4,2048,768]
  const float* W_attn = (const float*)d_in[1];   // [768,2304]
  const float* W_proj = (const float*)d_in[2];   // [768,768]
  const float* b_proj = (const float*)d_in[3];   // [768]
  float* out = (float*)d_out;

  char* ws = (char*)d_ws;
  f16* xb  = (f16*)(ws + 0);          // 8192*768    = 12,582,912 B
  f16* WaT = (f16*)(ws + 12582912);   // 2304*768    =  3,538,944 B
  f16* WpT = (f16*)(ws + 16121856);   // 768*768     =  1,179,648 B
  f16* qb  = (f16*)(ws + 17301504);   // 64*2048*64  = 16,777,216 B
  f16* kb  = (f16*)(ws + 34078720);   // 64*2048*64  = 16,777,216 B
  f16* vT  = (f16*)(ws + 50855936);   // 64*64*2048  = 16,777,216 B
  f16* ab  = (f16*)(ws + 67633152);   // 8192*768    = 12,582,912 B  (total ~80.2 MB)

  k_cvt_f16<<<2048, 256, 0, stream>>>(x, xb, 8192 * 768);
  k_transpose_f16<<<2048, 256, 0, stream>>>(W_attn, WaT, 768, 2304);
  k_transpose_f16<<<1024, 256, 0, stream>>>(W_proj, WpT, 768, 768);
  // zero q/k pads (whole q+k region, contiguous): 2*16,777,216 B = 8,388,608 dwords
  k_zero32<<<2048, 256, 0, stream>>>((unsigned int*)qb, 8388608);

  qkv_gemm<<<dim3(18, 64), 256, 0, stream>>>(xb, WaT, qb, kb, vT);
  attn_kernel<<<dim3(16, 64), 256, 0, stream>>>(qb, kb, vT, ab);
  proj_gemm<<<dim3(6, 64), 256, 0, stream>>>(ab, WpT, b_proj, out);
}
